// DriftingField_47442208752031
// MI455X (gfx1250) — matrix operand order repaired and checked by reading, not compiled, not run on hardware
//
#include <hip/hip_runtime.h>
#include <stdint.h>

// ---------------------------------------------------------------------------
// CDNA5 / gfx1250 wave32 implementation of the DriftingField reference.
//  * bf16x3 error-compensated WMMA GEMMs (v_wmma_f32_16x16x32_bf16)
//  * Tensor Data Mover (tensor_load_to_lds) double-buffered tile pipeline
//  * features pre-split into persistent bf16 hi/lo planes (L2-resident)
// ---------------------------------------------------------------------------

typedef __attribute__((ext_vector_type(16))) __bf16 v16bf;
typedef __attribute__((ext_vector_type(8)))  float  v8f;
typedef __attribute__((ext_vector_type(4)))  unsigned int v4u;
typedef __attribute__((ext_vector_type(8)))  int    v8i;
typedef __attribute__((ext_vector_type(4)))  int    v4i;

constexpr int G  = 4096;   // gen rows
constexpr int D  = 1024;   // feature dim
constexpr int T  = 8192;   // targets rows (gen ++ pos)
constexpr int TK = 32;     // WMMA K per step
constexpr int LDA = 40;    // LDS tile row pitch in bf16 (80B; TDM pad produces this)

constexpr float TEMP_INV = 20.0f;     // 1 / 0.05
constexpr float DNORM    = 0.03125f;  // 1 / sqrt(1024)

__device__ __forceinline__ v8f wmma_bf16(v16bf a, v16bf b, v8f c) {
  return __builtin_amdgcn_wmma_f32_16x16x32_bf16(false, a, false, b,
                                                 (short)0, c, false, false);
}

// A-matrix 16x32 bf16 fragment: row = lane&15; elem e -> k = e+8*half+(e>=8?8:0)
__device__ __forceinline__ v16bf frag_a(const __bf16* tile, int row, int half) {
  const __bf16* p = tile + row * LDA;
  v16bf f;
#pragma unroll
  for (int e = 0; e < 16; ++e) {
    int k = e + 8 * half + ((e >= 8) ? 8 : 0);
    f[e] = p[k];
  }
  return f;
}

// B-matrix 32x16 bf16 fragment (LDS stored [n][k]): elem e -> k = e+8*half+(e>=8?8:0)
__device__ __forceinline__ v16bf frag_b(const __bf16* tile, int nrow, int half) {
  const __bf16* p = tile + nrow * LDA;
  v16bf f;
#pragma unroll
  for (int e = 0; e < 16; ++e) f[e] = p[e + 8 * half + ((e >= 8) ? 8 : 0)];
  return f;
}

__device__ __forceinline__ void split_bf16(float v, __bf16& hi, __bf16& lo) {
  __bf16 h = (__bf16)v;
  hi = h;
  lo = (__bf16)(v - (float)h);
}

// ---------------------------------------------------------------------------
// Tensor Data Mover: DMA a 2-D bf16 tile (tile_d1 rows x tile_d0 cols) from a
// row-major [tensor_d1][tensor_d0] global tensor into LDS. LDS rows padded to
// 80B pitch (pad_interval = 16 dwords, pad_amount = 4 dwords) => LDA = 40.
// ---------------------------------------------------------------------------
__device__ __forceinline__ void tdm_load_2d_bf16(
    const __bf16* gsrc, void* lds_dst,
    unsigned tensor_d0, unsigned tensor_d1,
    unsigned tile_d0, unsigned tile_d1) {
  unsigned long long ga = (unsigned long long)(uintptr_t)gsrc;
  unsigned lds = (unsigned)(uintptr_t)lds_dst;
  v4u g0;
  g0[0] = 1u;                                             // count=1, user mode
  g0[1] = lds;                                            // lds_addr
  g0[2] = (unsigned)(ga & 0xFFFFFFFFu);                   // global_addr[31:0]
  g0[3] = (unsigned)((ga >> 32) & 0x01FFFFFFu) | (2u << 30);  // ga[56:32],type=2
  v8i g1;
  g1[0] = (int)((1u << 16)        // data_size = 2 bytes
              | (1u << 20)        // pad_enable
              | (3u << 22)        // pad_interval: 16 dwords (64B tile row)
              | (3u << 25));      // pad_amount : 4 dwords (+16B -> 80B pitch)
  g1[1] = (int)((tensor_d0 & 0xFFFFu) << 16);             // tensor_dim0 lo
  g1[2] = (int)(((tensor_d0 >> 16) & 0xFFFFu)
              | ((tensor_d1 & 0xFFFFu) << 16));           // dim0 hi | dim1 lo
  g1[3] = (int)(((tensor_d1 >> 16) & 0xFFFFu)
              | ((tile_d0 & 0xFFFFu) << 16));             // dim1 hi | tile_dim0
  g1[4] = (int)(tile_d1 & 0xFFFFu);                       // tile_dim1, tile_dim2=0
  g1[5] = (int)tensor_d0;                                 // tensor_dim0_stride lo
  g1[6] = 0;
  g1[7] = 0;
  v4i z = {0, 0, 0, 0};
#if defined(__clang_major__) && (__clang_major__ >= 23)
  v8i z8 = {0, 0, 0, 0, 0, 0, 0, 0};
  __builtin_amdgcn_tensor_load_to_lds(g0, g1, z, z, z8, 0);
#else
  __builtin_amdgcn_tensor_load_to_lds(g0, g1, z, z, 0);
#endif
}

// ---------------------------------------------------------------------------
// Stage 0a: squared row norms of all 8192 target rows.
// ---------------------------------------------------------------------------
__global__ __launch_bounds__(256) void norms_kernel(
    const float* __restrict__ gen, const float* __restrict__ pos,
    float* __restrict__ sqn) {
  __shared__ float red[256];
  const int row = blockIdx.x;
  const float* src = (row < G) ? (gen + (size_t)row * D)
                               : (pos + (size_t)(row - G) * D);
  float s = 0.0f;
  for (int c = threadIdx.x; c < D; c += 256) {
    float v = src[c];
    s += v * v;
  }
  red[threadIdx.x] = s;
  __syncthreads();
  for (int off = 128; off > 0; off >>= 1) {
    if (threadIdx.x < off) red[threadIdx.x] += red[threadIdx.x + off];
    __syncthreads();
  }
  if (threadIdx.x == 0) sqn[row] = red[0];
}

// ---------------------------------------------------------------------------
// Stage 0b: split features into persistent bf16 hi/lo planes [T][D].
// ---------------------------------------------------------------------------
__global__ __launch_bounds__(256) void convert_kernel(
    const float* __restrict__ gen, const float* __restrict__ pos,
    __bf16* __restrict__ thi, __bf16* __restrict__ tlo) {
  size_t idx = (size_t)blockIdx.x * 256 + threadIdx.x;
  const size_t halfN = (size_t)G * D;
  float v = (idx < halfN) ? gen[idx] : pos[idx - halfN];
  split_bf16(v, thi[idx], tlo[idx]);
}

// ---------------------------------------------------------------------------
// Stage 1: zero rowsum[G] ++ colsum[T] (contiguous).
// ---------------------------------------------------------------------------
__global__ __launch_bounds__(256) void zero_kernel(float* __restrict__ p, int n) {
  int i = blockIdx.x * 256 + threadIdx.x;
  if (i < n) p[i] = 0.0f;
}

// ---------------------------------------------------------------------------
// Stage 2: fused WMMA GEMM  S = gen @ targets^T -> dist -> exp -> K matrix.
// 256 thr (8 waves), block tile 64x128, each wave 16x64 (4 accumulators).
// TDM double-buffered pipeline: wave 0 waits prior DMA, issues next chunk,
// then all waves compute on the ready buffer.
// ---------------------------------------------------------------------------
__global__ __launch_bounds__(256) void dist_exp_kernel(
    const __bf16* __restrict__ thi, const __bf16* __restrict__ tlo,
    const float* __restrict__ sqn, float* __restrict__ Kmat,
    float* __restrict__ rowsum, float* __restrict__ colsum) {
  __shared__ __bf16 Ah[2][64 * LDA], Al[2][64 * LDA];
  __shared__ __bf16 Bh[2][128 * LDA], Bl[2][128 * LDA];
  __shared__ float rs_l[64], cs_l[128];
  __shared__ float s_sqi[64], s_sqj[128];

  const int j0  = blockIdx.x * 128;  // target column tile
  const int i0  = blockIdx.y * 64;   // gen row tile
  const int tid = threadIdx.x;

  if (tid < 64) {
    rs_l[tid] = 0.0f;
    s_sqi[tid] = sqn[i0 + tid];
  }
  if (tid < 128) {
    cs_l[tid] = 0.0f;
    s_sqj[tid] = sqn[j0 + tid];
  }

  v8f acc[4] = {};

  // Prologue: issue DMA for buffer 0 (TDM ignores EXEC; one wave issues).
  if (tid < 32) {
    tdm_load_2d_bf16(thi + (size_t)i0 * D, Ah[0], D, T, TK, 64);
    tdm_load_2d_bf16(tlo + (size_t)i0 * D, Al[0], D, T, TK, 64);
    tdm_load_2d_bf16(thi + (size_t)j0 * D, Bh[0], D, T, TK, 128);
    tdm_load_2d_bf16(tlo + (size_t)j0 * D, Bl[0], D, T, TK, 128);
  }

  int buf = 0;
  for (int kb = 0; kb < D; kb += TK) {
    if (tid < 32) __builtin_amdgcn_s_wait_tensorcnt(0);  // buf ready
    __syncthreads();  // all waves: buf visible; buf^1 reads retired
    if (tid < 32 && (kb + TK) < D) {                     // prefetch next chunk
      const int nk = kb + TK;
      const int nbuf = buf ^ 1;
      tdm_load_2d_bf16(thi + (size_t)i0 * D + nk, Ah[nbuf], D, T, TK, 64);
      tdm_load_2d_bf16(tlo + (size_t)i0 * D + nk, Al[nbuf], D, T, TK, 64);
      tdm_load_2d_bf16(thi + (size_t)j0 * D + nk, Bh[nbuf], D, T, TK, 128);
      tdm_load_2d_bf16(tlo + (size_t)j0 * D + nk, Bl[nbuf], D, T, TK, 128);
    }
    const int w = tid >> 5, lane = tid & 31;
    const int wm = w & 3, wn = w >> 2;       // 4 row-blocks x 2 col-blocks
    const int half = lane >> 4, ln = lane & 15;
    v16bf ah = frag_a(Ah[buf], wm * 16 + ln, half);
    v16bf al = frag_a(Al[buf], wm * 16 + ln, half);
#pragma unroll
    for (int nb = 0; nb < 4; ++nb) {
      int nrow = wn * 64 + nb * 16 + ln;
      v16bf bh = frag_b(Bh[buf], nrow, half);
      v16bf bl = frag_b(Bl[buf], nrow, half);
      acc[nb] = wmma_bf16(ah, bh, acc[nb]);   // hi*hi
      acc[nb] = wmma_bf16(ah, bl, acc[nb]);   // hi*lo
      acc[nb] = wmma_bf16(al, bh, acc[nb]);   // lo*hi
    }
    buf ^= 1;
  }
  __syncthreads();

  const int w = tid >> 5, lane = tid & 31;
  const int wm = w & 3, wn = w >> 2;
  const int half = lane >> 4, ln = lane & 15;
#pragma unroll
  for (int nb = 0; nb < 4; ++nb) {
    const int n_loc = wn * 64 + nb * 16 + ln;
    const int j = j0 + n_loc;
    const float sqj = s_sqj[n_loc];
#pragma unroll
    for (int r = 0; r < 8; ++r) {
      const int m_loc = wm * 16 + r + 8 * half;  // C/D layout: m = r + 8*half
      const int i = i0 + m_loc;
      float d2 = s_sqi[m_loc] + sqj - 2.0f * acc[nb][r];
      float dist = sqrtf(fmaxf(d2, 0.0f)) * DNORM;
      float kv = (j == i) ? 0.0f : __expf(-dist * TEMP_INV);  // diag -> 0
      Kmat[(size_t)i * T + j] = kv;
      atomicAdd(&rs_l[m_loc], kv);
      atomicAdd(&cs_l[n_loc], kv);
    }
  }
  __syncthreads();
  if (tid < 64)        atomicAdd(&rowsum[i0 + tid], rs_l[tid]);
  else if (tid < 192)  atomicAdd(&colsum[j0 + tid - 64], cs_l[tid - 64]);
}

// ---------------------------------------------------------------------------
// Stage 3: per-row half sums of normalized kernel (K matrix streamed from L2).
// ---------------------------------------------------------------------------
__global__ __launch_bounds__(256) void rowhalf_kernel(
    const float* __restrict__ Kmat, const float* __restrict__ rowsum,
    const float* __restrict__ colsum, float* __restrict__ rsg,
    float* __restrict__ rsp) {
  __shared__ float red_g[256], red_p[256];
  const int i = blockIdx.x;
  const float rs = rowsum[i];
  const float* krow = Kmat + (size_t)i * T;
  float sg = 0.0f, sp = 0.0f;
  for (int j = threadIdx.x; j < T; j += 256) {
    float nk = krow[j] * rsqrtf(fmaxf(rs * colsum[j], 1e-12f));
    if (j < G) sg += nk; else sp += nk;
  }
  red_g[threadIdx.x] = sg;
  red_p[threadIdx.x] = sp;
  __syncthreads();
  for (int off = 128; off > 0; off >>= 1) {
    if (threadIdx.x < off) {
      red_g[threadIdx.x] += red_g[threadIdx.x + off];
      red_p[threadIdx.x] += red_p[threadIdx.x + off];
    }
    __syncthreads();
  }
  if (threadIdx.x == 0) { rsg[i] = red_g[0]; rsp[i] = red_p[0]; }
}

// ---------------------------------------------------------------------------
// Stage 4: fused coefficient build + WMMA GEMM:
//   C[i,j] = (j>=G ? +nk*rsg[i] : -nk*rsp[i]),  out = C @ targets  [4096x1024]
// Block tile 64x128; A built by VALU from Kmat; B copied from bf16 planes.
// ---------------------------------------------------------------------------
__global__ __launch_bounds__(256) void coeff_gemm_kernel(
    const __bf16* __restrict__ thi, const __bf16* __restrict__ tlo,
    const float* __restrict__ Kmat, const float* __restrict__ rowsum,
    const float* __restrict__ colsum, const float* __restrict__ rsg,
    const float* __restrict__ rsp, float* __restrict__ out) {
  __shared__ __bf16 Ahi[64 * LDA], Alo[64 * LDA];
  __shared__ __bf16 Bhi[128 * LDA], Blo[128 * LDA];
  __shared__ float s_rs[64], s_g[64], s_p[64];

  const int n0  = blockIdx.x * 128;  // output column tile (feature dim)
  const int i0  = blockIdx.y * 64;   // output row tile
  const int tid = threadIdx.x;

  if (tid < 64) {
    s_rs[tid] = rowsum[i0 + tid];
    s_g[tid]  = rsg[i0 + tid];
    s_p[tid]  = rsp[i0 + tid];
  }

  v8f acc[4] = {};

  for (int kb = 0; kb < T; kb += TK) {
    __syncthreads();
    // A tile: 64 rows x 32 k of C, built on the fly from Kmat.
#pragma unroll
    for (int it = 0; it < 8; ++it) {
      int l = tid + it * 256;
      int r = l >> 5, c = l & 31;
      int j = kb + c;
      float kv = Kmat[(size_t)(i0 + r) * T + j];
      float nk = kv * rsqrtf(fmaxf(s_rs[r] * colsum[j], 1e-12f));
      float cc = (j >= G) ? nk * s_g[r] : -nk * s_p[r];
      split_bf16(cc, Ahi[r * LDA + c], Alo[r * LDA + c]);
    }
    // B tile: 32 k-rows x 128 cols from bf16 planes, stored transposed [n][k].
#pragma unroll
    for (int it = 0; it < 16; ++it) {
      int l = tid + it * 256;
      int k = l >> 7, n = l & 127;
      size_t src = (size_t)(kb + k) * D + n0 + n;
      Bhi[n * LDA + k] = thi[src];
      Blo[n * LDA + k] = tlo[src];
    }
    __syncthreads();
    if (kb + TK < T) {  // global_prefetch_b8 for next K-matrix chunk
      __builtin_prefetch(&Kmat[(size_t)(i0 + (tid >> 3)) * T + kb + TK], 0, 0);
    }
    const int w = tid >> 5, lane = tid & 31;
    const int wm = w & 3, wn = w >> 2;
    const int half = lane >> 4, ln = lane & 15;
    v16bf ah = frag_a(Ahi, wm * 16 + ln, half);
    v16bf al = frag_a(Alo, wm * 16 + ln, half);
#pragma unroll
    for (int nb = 0; nb < 4; ++nb) {
      int nrow = wn * 64 + nb * 16 + ln;
      v16bf bh = frag_b(Bhi, nrow, half);
      v16bf bl = frag_b(Blo, nrow, half);
      acc[nb] = wmma_bf16(ah, bh, acc[nb]);
      acc[nb] = wmma_bf16(ah, bl, acc[nb]);
      acc[nb] = wmma_bf16(al, bh, acc[nb]);
    }
  }

  const int w = tid >> 5, lane = tid & 31;
  const int wm = w & 3, wn = w >> 2;
  const int half = lane >> 4, ln = lane & 15;
#pragma unroll
  for (int nb = 0; nb < 4; ++nb) {
    const int n = n0 + wn * 64 + nb * 16 + ln;
#pragma unroll
    for (int r = 0; r < 8; ++r) {
      const int i = i0 + wm * 16 + r + 8 * half;
      out[(size_t)i * D + n] = acc[nb][r];
    }
  }
}

// ---------------------------------------------------------------------------
// Host entry. Workspace layout:
//   Kmat[G*T] f32 | rowsum[G] | colsum[T] | rsg[G] | rsp[G] | sqn[T] |
//   thi[T*D] bf16 | tlo[T*D] bf16                         (~160.2 MiB total)
// ---------------------------------------------------------------------------
extern "C" void kernel_launch(void* const* d_in, const int* in_sizes, int n_in,
                              void* d_out, int out_size, void* d_ws,
                              size_t ws_size, hipStream_t stream) {
  const float* gen = (const float*)d_in[0];
  const float* pos = (const float*)d_in[1];
  float* out = (float*)d_out;

  float* Kmat   = (float*)d_ws;
  float* rowsum = Kmat + (size_t)G * T;
  float* colsum = rowsum + G;
  float* rsg    = colsum + T;
  float* rsp    = rsg + G;
  float* sqn    = rsp + G;
  __bf16* thi   = (__bf16*)(sqn + T);
  __bf16* tlo   = thi + (size_t)T * D;

  norms_kernel<<<T, 256, 0, stream>>>(gen, pos, sqn);
  convert_kernel<<<(int)(((size_t)T * D) / 256), 256, 0, stream>>>(gen, pos,
                                                                   thi, tlo);
  zero_kernel<<<(G + T + 255) / 256, 256, 0, stream>>>(rowsum, G + T);
  dist_exp_kernel<<<dim3(T / 128, G / 64), 256, 0, stream>>>(thi, tlo, sqn,
                                                             Kmat, rowsum,
                                                             colsum);
  rowhalf_kernel<<<G, 256, 0, stream>>>(Kmat, rowsum, colsum, rsg, rsp);
  coeff_gemm_kernel<<<dim3(D / 128, G / 64), 256, 0, stream>>>(
      thi, tlo, Kmat, rowsum, colsum, rsg, rsp, out);
}
